// ConvNetShard1_38130719654466
// MI455X (gfx1250) — compile-verified, matrix operand order checked by reading, not run on hardware
//
#include <hip/hip_runtime.h>

// Conv2d(1->16,5x5,pad2) + bias + BN(inference) + ReLU + MaxPool2x2
// Implicit GEMM on v_wmma_f32_16x16x32_f16 (gfx1250, wave32).
//
// - Input staged in LDS pre-packed as u32 = (f16 x[c-2], f16 x[c-1]): every
//   A-fragment f16 pair is ONE aligned ds_load_b32. Fill is vectorized:
//   4 cells/task = 5 float loads + ds_store_b128 (edge tasks take slow path).
// - Window mapping with khalf-INVARIANT in-pair deltas {2, LDSW, 2, 2} ->
//   each A slot pair is one ds_load_2addr_b32 writing consecutive VGPRs of
//   the SAME fragment (verified: operands feed WMMA with no shuffle moves).
//   Windows with column >= 5 carry zero weights (padding reads real data).
// - B fragment (packed f16 weight pairs) + BN scale/shift built once per
//   block into LDS tables; lanes fetch with 2x ds_load_b128 / 1x ds_load_b64.
// - Each wave: 16 conv rows -> 8 pooled output rows (4 base VGPRs + DS
//   immediate offsets address all rows).
// - C layout (M=pixel in VGPR, N=channel in lane): 2x2 max-pool is pure
//   register ops; store is one global_store_b128 per output row per wave.
// - BN scale = gamma*rsqrt(var+eps) > 0, so affine+ReLU commute with max.

typedef __attribute__((ext_vector_type(16))) _Float16 v16h;
typedef __attribute__((ext_vector_type(8)))  float    v8f;
typedef __attribute__((ext_vector_type(4)))  float    v4f;
typedef __attribute__((ext_vector_type(4)))  uint32_t v4u;

#define IH   224
#define IW   224
#define OH   112
#define OW   112
#define COUT 16
#define LDSW 232          // packed u32 cells per row = 4*58 (b128-store exact)
#define NROW 20           // conv rows 16*ohq..16*ohq+15 need input rows -2..+17
#define NG   58           // 4-cell groups per row
#define NTHR 448          // 14 waves: one 16-wide conv segment each
#define ORPB 8            // pooled output rows per block

// window table: pair p -> (row KY[p], even column offset KC[p])
__device__ __constant__ const int KY_tab[16] = {0,0,0,1, 1,1,2,3, 2,2,4,4, 3,3,4,4};
__device__ __constant__ const int KC_tab[16] = {0,2,4,4, 0,2,0,0, 2,4,0,2, 2,4,4,6};

union AFrag { uint32_t u[8]; v16h h; };
union BFrag { v4u q[2];     v16h h; };

__device__ __forceinline__ uint32_t pk2(float a, float b) {
    const uint32_t ua = (uint32_t)__builtin_bit_cast(uint16_t, (_Float16)a);
    const uint32_t ub = (uint32_t)__builtin_bit_cast(uint16_t, (_Float16)b);
    return ua | (ub << 16);
}

__global__ __launch_bounds__(NTHR)
void conv_bn_relu_pool_wmma(const float* __restrict__ x,
                            const float* __restrict__ conv_w,
                            const float* __restrict__ conv_b,
                            const float* __restrict__ gamma,
                            const float* __restrict__ beta,
                            const float* __restrict__ run_mean,
                            const float* __restrict__ run_var,
                            float* __restrict__ out)
{
    __shared__ uint32_t P[NROW * LDSW];   // packed input pairs
    __shared__ uint32_t BW[256];          // [khalf][n][v] packed f16 weight pairs
    __shared__ float    BNT[2 * COUT];    // per-channel {scale, shift}

    const int ohq  = blockIdx.x;          // group of 8 output rows
    const int b    = blockIdx.y;
    const int tid  = threadIdx.x;
    const int lane = tid & 31;
    const int wave = tid >> 5;            // 0..13 -> 16-column conv segment

    // ---- stage 20 input rows (16*ohq-2 .. 16*ohq+17) as packed f16 pairs ---
    // task = 4 consecutive cells; cell cx packs (x[cx-2], x[cx-1]).
    const float* xb = x + (size_t)b * IH * IW;
    for (int t = tid; t < NROW * NG; t += NTHR) {
        const int ry  = t / NG;
        const int cx0 = (t - ry * NG) * 4;
        const int ir  = 16 * ohq - 2 + ry;
        uint32_t pk[4];
        if (ir >= 0 && ir < IH && cx0 >= 4 && cx0 <= 220) {
            // interior fast path: 5 consecutive floats
            const float* r = xb + ir * IW + (cx0 - 2);
            const float f0 = r[0], f1 = r[1], f2 = r[2], f3 = r[3], f4 = r[4];
            pk[0] = pk2(f0, f1); pk[1] = pk2(f1, f2);
            pk[2] = pk2(f2, f3); pk[3] = pk2(f3, f4);
        } else {
            #pragma unroll
            for (int k = 0; k < 4; ++k) {
                const int cl = cx0 + k - 2, ch = cx0 + k - 1;
                float lo = 0.0f, hi = 0.0f;
                if (ir >= 0 && ir < IH) {
                    if (cl >= 0 && cl < IW) lo = xb[ir * IW + cl];
                    if (ch >= 0 && ch < IW) hi = xb[ir * IW + ch];
                }
                pk[k] = pk2(lo, hi);
            }
        }
        *(v4u*)&P[ry * LDSW + cx0] = *(v4u*)pk;   // ds_store_b128 (16B aligned)
    }

    // ---- build packed B-weight table once per block: BW[h*128 + n*8 + v] ---
    if (tid < 256) {
        const int h = tid >> 7, v = tid & 7, nn = (tid >> 3) & 15;
        const int p = (v < 4) ? h * 4 + v : 8 + h * 4 + (v - 4);
        const int ky = KY_tab[p], c = KC_tab[p];
        float w0 = 0.0f, w1 = 0.0f;
        if (c     < 5) w0 = conv_w[nn * 25 + ky * 5 + c];
        if (c + 1 < 5) w1 = conv_w[nn * 25 + ky * 5 + c + 1];
        BW[tid] = pk2(w0, w1);
    }
    // ---- BN scale/shift table once per block -------------------------------
    if (tid < COUT) {
        const float sc = gamma[tid] * rsqrtf(run_var[tid] + 1e-5f);
        BNT[2 * tid]     = sc;
        BNT[2 * tid + 1] = beta[tid] + (conv_b[tid] - run_mean[tid]) * sc;
    }

    __syncthreads();

    // ---- per-lane setup ----------------------------------------------------
    const int n     = lane & 15;          // output channel (B/C column)
    const int khalf = lane >> 4;
    const int m     = lane & 15;          // pixel within segment (A row)

    BFrag bf;                              // 2x ds_load_b128 (32B aligned)
    const v4u* bwp = (const v4u*)&BW[(khalf * 16 + n) * 8];
    bf.q[0] = bwp[0];
    bf.q[1] = bwp[1];
    const float2 ss = *(const float2*)&BNT[2 * n];   // ds_load_b64

    // 4 shared base addresses (element units); in-pair deltas are literals.
    const int baseCol = wave * 16 + m;
    int bj[4];
    bj[0] = baseCol + (khalf ? 1 * LDSW     : 0);            // (ky1,c0)/(ky0,c0)
    bj[1] = baseCol + (khalf ? 2 * LDSW     : 4);            // (ky2,c0)/(ky0,c4)
    bj[2] = baseCol + (khalf ? 3 * LDSW + 2 : 2 * LDSW + 2); // (ky3,c2)/(ky2,c2)
    bj[3] = baseCol + (khalf ? 4 * LDSW + 4 : 4 * LDSW);     // (ky4,c4)/(ky4,c0)
    const int DJ[4] = {2, LDSW, 2, 2};

    // ---- 16 conv rows -> 8 pooled output rows ------------------------------
    const int ow = wave * 8 + (khalf << 2);
    float* op = out + (((size_t)(b * COUT + n) * OH + ORPB * ohq) * OW + ow);

    #pragma unroll
    for (int rp = 0; rp < ORPB; ++rp) {
        v8f acc[2];
        #pragma unroll
        for (int h2 = 0; h2 < 2; ++h2) {
            const int cr = 2 * rp + h2;    // conv row within block (0..15)
            AFrag a;
            #pragma unroll
            for (int j = 0; j < 4; ++j) {  // one ds_load_2addr_b32 per pair
                a.u[2 * j]     = P[bj[j] + cr * LDSW];
                a.u[2 * j + 1] = P[bj[j] + cr * LDSW + DJ[j]];
            }
            v8f c = {};
            acc[h2] = __builtin_amdgcn_wmma_f32_16x16x32_f16(
                          false, a.h, false, bf.h, (short)0, c, false, false);
        }

        // 2x2 max-pool in registers, then BN affine + ReLU (scale > 0)
        v4f pl;
        #pragma unroll
        for (int i = 0; i < 4; ++i) {
            const float v0 = fmaxf(acc[0][2 * i],     acc[1][2 * i]);
            const float v1 = fmaxf(acc[0][2 * i + 1], acc[1][2 * i + 1]);
            pl[i] = fmaxf(fmaxf(v0, v1) * ss.x + ss.y, 0.0f);
        }

        *(v4f*)(op + (size_t)rp * OW) = pl;   // global_store_b128
    }
}

extern "C" void kernel_launch(void* const* d_in, const int* in_sizes, int n_in,
                              void* d_out, int out_size, void* d_ws, size_t ws_size,
                              hipStream_t stream) {
    const float* x      = (const float*)d_in[0];
    const float* conv_w = (const float*)d_in[1];
    const float* conv_b = (const float*)d_in[2];
    const float* gamma  = (const float*)d_in[3];
    const float* beta   = (const float*)d_in[4];
    const float* rmean  = (const float*)d_in[5];
    const float* rvar   = (const float*)d_in[6];

    const int B = in_sizes[0] / (IH * IW);   // 128
    dim3 grid(OH / ORPB, B);                 // one block per (batch, 8 output rows)
    conv_bn_relu_pool_wmma<<<grid, NTHR, 0, stream>>>(
        x, conv_w, conv_b, gamma, beta, rmean, rvar, (float*)d_out);
}
